// VerticalNLBlock_76020921139498
// MI455X (gfx1250) — compile-verified, matrix operand order checked by reading, not run on hardware
//
#include <hip/hip_runtime.h>

// VerticalNLBlock: per-(b,w) self-attention over [h=128, c=64] slices.
// x: [8, 64, 128, 256] f32. 2048 independent problems, one per workgroup.

typedef __bf16 bf16_t;
typedef bf16_t v4bf  __attribute__((ext_vector_type(4)));
typedef bf16_t v8bf  __attribute__((ext_vector_type(8)));
typedef bf16_t v16bf __attribute__((ext_vector_type(16)));
typedef float  v4f   __attribute__((ext_vector_type(4)));
typedef float  v8f   __attribute__((ext_vector_type(8)));

#define DIM_B 8
#define DIM_C 64
#define DIM_H 128
#define DIM_W 256

union BfPack { v16bf v; v8bf h[2]; };

// Build a 16x32 bf16 WMMA operand from a row-major LDS row: two contiguous
// 16B chunks at k0 and k0+16 (CDNA5 16-bit A/B VGPR layout: lanes 0-15 hold
// K {0..7,16..23}, lanes 16-31 hold K {8..15,24..31}).
__device__ __forceinline__ v16bf ld_op(const bf16_t* rowBase, int k0) {
  BfPack u;
  u.h[0] = *(const v8bf*)(rowBase + k0);
  u.h[1] = *(const v8bf*)(rowBase + k0 + 16);
  return u.v;
}

__global__ __launch_bounds__(256) void vnl_kernel(const float* __restrict__ x,
                                                  float* __restrict__ out) {
  extern __shared__ char smem[];
  float*  sc   = (float*)(smem);            // [128][128] f32 scores, 64KB
  bf16_t* Abf  = (bf16_t*)(smem + 65536);   // [128][128] bf16 exp(s-M), 32KB
  bf16_t* Xbf  = (bf16_t*)(smem + 98304);   // [h=128][c=64] bf16, 16KB
  bf16_t* XbfT = (bf16_t*)(smem + 114688);  // [c=64][h=128] bf16, 16KB
  float*  red8 = (float*)(smem + 131072);   // [8][128] f32 partials, 4KB
  float*  Mbuf = (float*)(smem + 135168);   // [128] column max
  float*  Sbuf = (float*)(smem + 135680);   // [128] column 1/sum

  const int tid = threadIdx.x;
  const int bb  = blockIdx.x >> 8;   // batch
  const int w   = blockIdx.x & 255;  // column

  // ---- Phase 1: gather X (stride-w) -> bf16 row-major + transposed in LDS.
  {
    const size_t base = ((size_t)bb * DIM_C) * DIM_H * DIM_W + w;
#pragma unroll
    for (int i = 0; i < 32; ++i) {
      int idx = tid + i * 256;         // 0..8191 over (c,h)
      int c = idx >> 7;
      int h = idx & 127;
      float v = x[base + ((size_t)c * DIM_H + h) * DIM_W];
      bf16_t bv = (bf16_t)v;
      Xbf[h * DIM_C + c]  = bv;
      XbfT[c * DIM_H + h] = bv;
    }
  }
  __syncthreads();

  const int wave  = tid >> 5;      // 8 waves
  const int lane  = tid & 31;
  const int lrow  = lane & 15;     // M (A) / N (B) index within tile
  const int khalf = lane >> 4;     // which K half this lane holds
  const int kbase = khalf * 8;
  const int m0    = wave * 16;     // this wave's M strip

  // ---- Phase 2: scores[h][k] = sum_c X[h,c]*X[k,c]; wave -> 16x128 strip.
  {
    const bf16_t* arow = Xbf + (m0 + lrow) * DIM_C;
    const v16bf a0 = ld_op(arow, kbase);        // invariant across N tiles
    const v16bf a1 = ld_op(arow, 32 + kbase);
#pragma unroll
    for (int n = 0; n < 8; ++n) {
      const bf16_t* brow = Xbf + (n * 16 + lrow) * DIM_C;
      v16bf b0 = ld_op(brow, kbase);
      v16bf b1 = ld_op(brow, 32 + kbase);
      v8f acc = {};
      acc = __builtin_amdgcn_wmma_f32_16x16x32_bf16(
          false, a0, false, b0, (short)0, acc, false, false);
      acc = __builtin_amdgcn_wmma_f32_16x16x32_bf16(
          false, a1, false, b1, (short)0, acc, false, false);
      // C/D layout: lane l, VGPR v -> (M = v + 8*(l>>4), N = l&15)
#pragma unroll
      for (int v = 0; v < 8; ++v)
        sc[(m0 + v + 8 * khalf) * DIM_H + n * 16 + lrow] = acc[v];
    }
  }
  __syncthreads();

  // ---- Phase 3: column-wise softmax (over h), fully vectorized.
  // Each thread owns 4 consecutive columns x 16 rows: b128 loads, 4 chains.
  {
    const int cg    = tid & 31;    // column group: columns cg*4 .. cg*4+3
    const int rg    = tid >> 5;    // row group: rows rg*16 .. rg*16+15
    const int c4    = cg * 4;
    const int rbase = rg * 16;

    // Pass 1: partial max over 16 rows for 4 columns.
    v4f mx = {-3.402823466e+38f, -3.402823466e+38f,
              -3.402823466e+38f, -3.402823466e+38f};
#pragma unroll
    for (int r = 0; r < 16; ++r) {
      v4f v = *(const v4f*)(sc + (rbase + r) * DIM_H + c4);
#pragma unroll
      for (int j = 0; j < 4; ++j) mx[j] = fmaxf(mx[j], v[j]);
    }
    *(v4f*)(red8 + rg * 128 + c4) = mx;
    __syncthreads();
    if (tid < 128) {
      float m = red8[tid];
#pragma unroll
      for (int g = 1; g < 8; ++g) m = fmaxf(m, red8[g * 128 + tid]);
      Mbuf[tid] = m;
    }
    __syncthreads();

    // Pass 2: e = exp(s - M): partial sums + unnormalized bf16 A (b64 store).
    const v4f Mv = *(const v4f*)(Mbuf + c4);
    v4f sum = {0.f, 0.f, 0.f, 0.f};
#pragma unroll
    for (int r = 0; r < 16; ++r) {
      v4f v = *(const v4f*)(sc + (rbase + r) * DIM_H + c4);
      v4bf eb;
#pragma unroll
      for (int j = 0; j < 4; ++j) {
        float e = __expf(v[j] - Mv[j]);
        sum[j] += e;
        eb[j] = (bf16_t)e;
      }
      *(v4bf*)(Abf + (rbase + r) * DIM_H + c4) = eb;
    }
    *(v4f*)(red8 + rg * 128 + c4) = sum;
    __syncthreads();
    if (tid < 128) {
      float s = red8[tid];
#pragma unroll
      for (int g = 1; g < 8; ++g) s += red8[g * 128 + tid];
      Sbuf[tid] = 1.0f / s;     // invS[k]
    }
    __syncthreads();
  }

  // ---- Phase 3b: fold invS[k] into XbfT (out[h,c] = sum_k e[h,k]*X[k,c]/S[k]).
  // Thread owns 32 contiguous elements -> 4x (b128 load + b128 store).
  {
    const int base = tid * 32;        // c = base>>7, k0 = base&127
    const int k0   = base & 127;
#pragma unroll
    for (int j = 0; j < 4; ++j) {
      v8bf xv = *(const v8bf*)(XbfT + base + j * 8);
      v4f s0 = *(const v4f*)(Sbuf + k0 + j * 8);
      v4f s1 = *(const v4f*)(Sbuf + k0 + j * 8 + 4);
      v8bf rv;
#pragma unroll
      for (int e = 0; e < 4; ++e) rv[e] = (bf16_t)((float)xv[e] * s0[e]);
#pragma unroll
      for (int e = 0; e < 4; ++e) rv[4 + e] = (bf16_t)((float)xv[4 + e] * s1[e]);
      *(v8bf*)(XbfT + base + j * 8) = rv;
    }
  }
  __syncthreads();

  // ---- Phase 4: out[h][c] = sum_k e[h,k] * Xs[k,c]; write as [b,c,h,w].
  {
    const bf16_t* arow  = Abf + (m0 + lrow) * DIM_H;
    const size_t  gbase = ((size_t)bb * DIM_C) * DIM_H * DIM_W + w;
    const v16bf a0 = ld_op(arow, kbase);        // invariant across N tiles
    const v16bf a1 = ld_op(arow, 32 + kbase);
    const v16bf a2 = ld_op(arow, 64 + kbase);
    const v16bf a3 = ld_op(arow, 96 + kbase);
#pragma unroll
    for (int n = 0; n < 4; ++n) {
      const bf16_t* brow = XbfT + (n * 16 + lrow) * DIM_H;  // B as [N=c][K=h]
      v16bf b0 = ld_op(brow, kbase);
      v16bf b1 = ld_op(brow, 32 + kbase);
      v16bf b2 = ld_op(brow, 64 + kbase);
      v16bf b3 = ld_op(brow, 96 + kbase);
      v8f acc = {};
      acc = __builtin_amdgcn_wmma_f32_16x16x32_bf16(
          false, a0, false, b0, (short)0, acc, false, false);
      acc = __builtin_amdgcn_wmma_f32_16x16x32_bf16(
          false, a1, false, b1, (short)0, acc, false, false);
      acc = __builtin_amdgcn_wmma_f32_16x16x32_bf16(
          false, a2, false, b2, (short)0, acc, false, false);
      acc = __builtin_amdgcn_wmma_f32_16x16x32_bf16(
          false, a3, false, b3, (short)0, acc, false, false);
      const int c = n * 16 + lrow;
#pragma unroll
      for (int v = 0; v < 8; ++v) {
        const int h = m0 + v + 8 * khalf;
        out[gbase + ((size_t)c * DIM_H + h) * DIM_W] = acc[v];
      }
    }
  }
}

extern "C" void kernel_launch(void* const* d_in, const int* in_sizes, int n_in,
                              void* d_out, int out_size, void* d_ws, size_t ws_size,
                              hipStream_t stream) {
  (void)in_sizes; (void)n_in; (void)d_ws; (void)ws_size; (void)out_size;
  const float* x = (const float*)d_in[0];
  float* out = (float*)d_out;
  dim3 grid(DIM_B * DIM_W);   // 2048 workgroups, one (b,w) problem each
  dim3 block(256);            // 8 wave32
  const size_t shmem = 136192; // sc 64K + A 32K + X 16K + X^T 16K + scratch
  vnl_kernel<<<grid, block, shmem, stream>>>(x, out);
}
